// genecell_nn_77232101916748
// MI455X (gfx1250) — matrix operand order repaired and checked by reading, not compiled, not run on hardware
//
#include <hip/hip_runtime.h>
#include <hip/hip_bf16.h>
#include <math.h>

typedef __attribute__((ext_vector_type(16))) __bf16    v16bf;
typedef __attribute__((ext_vector_type(8)))  float     v8f;
typedef __attribute__((ext_vector_type(4)))  unsigned  v4u;
typedef __attribute__((ext_vector_type(8)))  int       v8i;
typedef __attribute__((ext_vector_type(4)))  int       v4i;
typedef unsigned short ushort_t;

#define BATCH   4096
#define NCOLS   1536     // 64 terms * 24 (GD)
#define KDIM    4096
#define TILE_M  128
#define TILE_N  128
#define TILE_K  64
#define LDSP    72       // padded LDS pitch in bf16 elems (144B rows, conflict-free)
#define NBLK    16       // BATCH / 256

#if __has_builtin(__builtin_amdgcn_tensor_load_to_lds) && __has_builtin(__builtin_amdgcn_s_wait_tensorcnt)
#define HAVE_TDM 1
#endif

__device__ __forceinline__ __bf16 f2bf(float f) {
  union { float f; unsigned u; } c; c.f = f;
  unsigned r = c.u + 0x7FFFu + ((c.u >> 16) & 1u);   // round-to-nearest-even
  union { unsigned short s; __bf16 b; } o;
  o.s = (unsigned short)(r >> 16);
  return o.b;
}

__device__ __forceinline__ unsigned pk2bf(float lo, float hi) {
  union { float f; unsigned u; } a, b; a.f = lo; b.f = hi;
  unsigned ra = (a.u + 0x7FFFu + ((a.u >> 16) & 1u)) >> 16;
  unsigned rb = (b.u + 0x7FFFu + ((b.u >> 16) & 1u)) >> 16;
  return (ra & 0xFFFFu) | (rb << 16);
}

// ---------------------------------------------------------------------------
// f32 -> bf16 pre-conversion (pure bandwidth pass). One thread = 8 elements.
// ---------------------------------------------------------------------------
__global__ __launch_bounds__(256)
void cvt_f32_bf16_kernel(const float* __restrict__ s, unsigned* __restrict__ d, int n8)
{
  int i = blockIdx.x * 256 + threadIdx.x;
  if (i >= n8) return;
  const float4* sp = (const float4*)s + (size_t)i * 2;
  float4 a = sp[0], b = sp[1];
  uint4 o;
  o.x = pk2bf(a.x, a.y); o.y = pk2bf(a.z, a.w);
  o.z = pk2bf(b.x, b.y); o.w = pk2bf(b.z, b.w);
  *((uint4*)d + i) = o;
}

// ---------------------------------------------------------------------------
// Shared WMMA inner compute: 4x2 subtiles of 16x16 per wave, two K=32 steps.
// Fragment layouts per CDNA5 ISA (A 16x32 M-major, B 32x16 across lanes).
// ---------------------------------------------------------------------------
__device__ __forceinline__ void wmma_tile_compute(
    const __bf16* As, const __bf16* Bs, v8f acc[4][2],
    int wm, int wn, int ln, int lh)
{
#pragma unroll
  for (int kk = 0; kk < TILE_K; kk += 32) {
    v16bf afr[4], bfr[2];
#pragma unroll
    for (int mt = 0; mt < 4; ++mt) {
      const unsigned* s32 = (const unsigned*)(As + (wm + mt * 16 + ln) * LDSP + kk);
      union { unsigned u[8]; v16bf v; } fa;
#pragma unroll
      for (int p = 0; p < 4; ++p) {
        fa.u[p]     = s32[lh * 4 + p];
        fa.u[4 + p] = s32[8 + lh * 4 + p];
      }
      afr[mt] = fa.v;
    }
#pragma unroll
    for (int nt = 0; nt < 2; ++nt) {
      const unsigned* s32 = (const unsigned*)(Bs + (wn + nt * 16 + ln) * LDSP + kk);
      union { unsigned u[8]; v16bf v; } fb;
#pragma unroll
      for (int p = 0; p < 4; ++p) {
        fb.u[p]     = s32[lh * 4 + p];
        fb.u[4 + p] = s32[8 + lh * 4 + p];
      }
      bfr[nt] = fb.v;
    }
#pragma unroll
    for (int mt = 0; mt < 4; ++mt)
#pragma unroll
      for (int nt = 0; nt < 2; ++nt)
        acc[mt][nt] = __builtin_amdgcn_wmma_f32_16x16x32_bf16(
            false, afr[mt], false, bfr[nt], (short)0, acc[mt][nt], false, false);
  }
}

__device__ __forceinline__ void wmma_store_c(
    v8f acc[4][2], const float* __restrict__ bias, float* __restrict__ C,
    size_t bm, size_t bn, int wm, int wn, int ln, int lh)
{
#pragma unroll
  for (int mt = 0; mt < 4; ++mt)
#pragma unroll
    for (int nt = 0; nt < 2; ++nt) {
      size_t col = bn + wn + nt * 16 + ln;
      float bv = bias[col];
#pragma unroll
      for (int p = 0; p < 8; ++p) {
        size_t row = bm + wm + mt * 16 + p + lh * 8;
        C[row * (size_t)NCOLS + col] = acc[mt][nt][p] + bv;
      }
    }
}

#ifdef HAVE_TDM
// Build a 2D tile descriptor (tile 64 x 128 of 2B elems, row stride 4096 elems,
// LDS padding 16B per 128B row -> 144B pitch) and issue TENSOR_LOAD_TO_LDS.
// This toolchain's builtin takes 6 args: g0, g1, g2, g3, extra group, cpol.
__device__ __forceinline__ void tdm_load_tile(unsigned long long gaddr,
                                              unsigned lds_addr, unsigned tdim1)
{
  v4u g0;
  g0.x = 1u;                                        // count=1, user descriptor
  g0.y = lds_addr;                                  // LDS byte address
  g0.z = (unsigned)(gaddr & 0xFFFFFFFFull);         // global addr [31:0]
  g0.w = (unsigned)((gaddr >> 32) & 0x1FFFFFFu) | (2u << 30);  // [56:32] | type=2
  v8i g1;
  // data_size=2B | pad_enable | pad_interval=4 (32 DW = 128B) | pad_amount=3 (4 DW = 16B)
  g1[0] = (int)((1u << 16) | (1u << 20) | (4u << 22) | (3u << 25));
  g1[1] = (int)((unsigned)(KDIM & 0xFFFF) << 16);   // tensor_dim0[15:0]
  g1[2] = (int)(((unsigned)KDIM >> 16) | ((tdim1 & 0xFFFFu) << 16)); // dim0 hi | dim1 lo
  g1[3] = (int)((tdim1 >> 16) | ((unsigned)TILE_K << 16));           // dim1 hi | tile_dim0
  g1[4] = (int)((unsigned)TILE_M);                  // tile_dim1=128, tile_dim2=0
  g1[5] = (int)((unsigned)KDIM);                    // tensor_dim0_stride lo32
  g1[6] = 0;
  g1[7] = 0;
  v4i z4 = {0, 0, 0, 0};
  v8i z8 = {0, 0, 0, 0, 0, 0, 0, 0};
  __builtin_amdgcn_tensor_load_to_lds(g0, g1, z4, z4, z8, 0);
}
#endif

// ---------------------------------------------------------------------------
// bf16-fed GEMM: C[b,n] = sum_g Xb[b,g]*Wb[n,g] + bias[n].
// TDM double-buffered DMA when available; manual B128 staging otherwise.
// ---------------------------------------------------------------------------
__global__ __launch_bounds__(256)
void gemm_bf16_kernel(const ushort_t* __restrict__ Xb, const ushort_t* __restrict__ Wb,
                      const float* __restrict__ bias, float* __restrict__ C)
{
  __shared__ __bf16 As[2][TILE_M * LDSP];
  __shared__ __bf16 Bs[2][TILE_N * LDSP];

  const int tid  = threadIdx.x;
  const int lane = tid & 31;
  const int wv   = tid >> 5;
  const int wm   = (wv >> 2) * 64;
  const int wn   = (wv & 3) * 32;
  const int ln   = lane & 15;
  const int lh   = lane >> 4;
  const size_t bm = (size_t)blockIdx.x * TILE_M;
  const size_t bn = (size_t)blockIdx.y * TILE_N;

  v8f acc[4][2];
#pragma unroll
  for (int i = 0; i < 4; ++i)
#pragma unroll
    for (int j = 0; j < 2; ++j) acc[i][j] = {};

#ifdef HAVE_TDM
  const unsigned long long xg = (unsigned long long)(size_t)Xb + bm * (size_t)KDIM * 2;
  const unsigned long long wg = (unsigned long long)(size_t)Wb + bn * (size_t)KDIM * 2;
  const unsigned asb[2] = { (unsigned)(size_t)&As[0][0], (unsigned)(size_t)&As[1][0] };
  const unsigned bsb[2] = { (unsigned)(size_t)&Bs[0][0], (unsigned)(size_t)&Bs[1][0] };
  const int NT_ = KDIM / TILE_K;

  if (wv == 0) {                       // prologue: DMA tile 0 into buffer 0
    tdm_load_tile(xg, asb[0], (unsigned)BATCH);
    tdm_load_tile(wg, bsb[0], (unsigned)NCOLS);
  }
  for (int it = 0; it < NT_; ++it) {
    const int buf = it & 1;
    if (wv == 0) {
      if (it + 1 < NT_) {              // prefetch tile it+1 into other buffer
        unsigned long long ko = (unsigned long long)(it + 1) * TILE_K * 2ull;
        tdm_load_tile(xg + ko, asb[buf ^ 1], (unsigned)BATCH);
        tdm_load_tile(wg + ko, bsb[buf ^ 1], (unsigned)NCOLS);
        __builtin_amdgcn_s_wait_tensorcnt(2);   // tile `it` has landed
      } else {
        __builtin_amdgcn_s_wait_tensorcnt(0);
      }
    }
    __syncthreads();                   // publish tile `it` to all waves
    wmma_tile_compute(As[buf], Bs[buf], acc, wm, wn, ln, lh);
    __syncthreads();                   // buffer free for next DMA
  }
#else
  for (int k0 = 0; k0 < KDIM; k0 += TILE_K) {
    // 1024 16B chunks per matrix, 4 per thread.
#pragma unroll
    for (int i = 0; i < 4; ++i) {
      int idx = tid + i * 256;
      int row = idx >> 3;
      int c   = (idx & 7) << 3;        // element offset within row
      *(uint4*)(&As[0][row * LDSP + c]) =
          *(const uint4*)(Xb + (bm + row) * (size_t)KDIM + k0 + c);
      *(uint4*)(&Bs[0][row * LDSP + c]) =
          *(const uint4*)(Wb + (bn + row) * (size_t)KDIM + k0 + c);
    }
    __syncthreads();
    wmma_tile_compute(As[0], Bs[0], acc, wm, wn, ln, lh);
    __syncthreads();
  }
#endif
  wmma_store_c(acc, bias, C, bm, bn, wm, wn, ln, lh);
}

// ---------------------------------------------------------------------------
// Fallback GEMM reading f32 directly (inline convert) — used if ws too small.
// ---------------------------------------------------------------------------
__global__ __launch_bounds__(256)
void gemm_xw_kernel(const float* __restrict__ X, const float* __restrict__ W,
                    const float* __restrict__ bias, float* __restrict__ C)
{
  __shared__ __bf16 As[TILE_M * LDSP];
  __shared__ __bf16 Bs[TILE_N * LDSP];

  const int tid  = threadIdx.x;
  const int lane = tid & 31;
  const int wv   = tid >> 5;
  const int wm   = (wv >> 2) * 64;
  const int wn   = (wv & 3) * 32;
  const int ln   = lane & 15;
  const int lh   = lane >> 4;
  const size_t bm = (size_t)blockIdx.x * TILE_M;
  const size_t bn = (size_t)blockIdx.y * TILE_N;

  v8f acc[4][2];
#pragma unroll
  for (int i = 0; i < 4; ++i)
#pragma unroll
    for (int j = 0; j < 2; ++j) acc[i][j] = {};

  for (int k0 = 0; k0 < KDIM; k0 += TILE_K) {
#pragma unroll
    for (int i = 0; i < 8; ++i) {
      int idx = tid + i * 256;
      int row = idx >> 4;
      int c4  = (idx & 15) << 2;
      float4 vx = *(const float4*)(X + (bm + row) * (size_t)KDIM + k0 + c4);
      __bf16* da = As + row * LDSP + c4;
      da[0] = f2bf(vx.x); da[1] = f2bf(vx.y); da[2] = f2bf(vx.z); da[3] = f2bf(vx.w);
      float4 vw = *(const float4*)(W + (bn + row) * (size_t)KDIM + k0 + c4);
      __bf16* db = Bs + row * LDSP + c4;
      db[0] = f2bf(vw.x); db[1] = f2bf(vw.y); db[2] = f2bf(vw.z); db[3] = f2bf(vw.w);
    }
    __syncthreads();
    wmma_tile_compute(As, Bs, acc, wm, wn, ln, lh);
    __syncthreads();
  }
  wmma_store_c(acc, bias, C, bm, bn, wm, wn, ln, lh);
}

// ---------------------------------------------------------------------------
// Per-level: z = tanh(in @ term_W^T + b) and per-block BN partial sums.
// ---------------------------------------------------------------------------
__global__ __launch_bounds__(256)
void level_z_kernel(const float* __restrict__ in, const float* __restrict__ tW,
                    const float* __restrict__ tb, float* __restrict__ z,
                    float* __restrict__ part, int n, int nblk)
{
  const int t   = blockIdx.x;
  const int tid = threadIdx.x;
  const int b   = blockIdx.y * 256 + tid;

  __shared__ float sW[6 * 24];
  __shared__ float sb[6];
  __shared__ float red[12 * 256];

  if (tid < 144) sW[tid] = tW[t * 144 + tid];
  if (tid < 6)   sb[tid] = tb[t * 6 + tid];
  __syncthreads();

  const float* src = in + ((size_t)b * n + t) * 24;
  float xin[24];
#pragma unroll
  for (int i = 0; i < 24; ++i) xin[i] = src[i];

  float* zo = z + ((size_t)b * n + t) * 6;
#pragma unroll
  for (int hh = 0; hh < 6; ++hh) {
    float s = sb[hh];
#pragma unroll
    for (int i = 0; i < 24; ++i) s += xin[i] * sW[hh * 24 + i];
    float zz = tanhf(s);
    zo[hh] = zz;
    red[hh * 256 + tid]       = zz;
    red[(6 + hh) * 256 + tid] = zz * zz;
  }
  __syncthreads();
  for (int s = 128; s > 0; s >>= 1) {
    if (tid < s)
#pragma unroll
      for (int a = 0; a < 12; ++a) red[a * 256 + tid] += red[a * 256 + tid + s];
    __syncthreads();
  }
  if (tid < 12) part[((size_t)t * 12 + tid) * nblk + blockIdx.y] = red[tid * 256];
}

__global__ void level_stats_kernel(const float* __restrict__ part,
                                   const float* __restrict__ gamma,
                                   const float* __restrict__ beta,
                                   float* __restrict__ stats,
                                   int n, int nblk, int btot)
{
  int idx = blockIdx.x * blockDim.x + threadIdx.x;   // t*6 + hh
  if (idx >= n * 6) return;
  int t = idx / 6, hh = idx % 6;
  float s = 0.f, sq = 0.f;
  for (int k = 0; k < nblk; ++k) {
    s  += part[((size_t)t * 12 + hh) * nblk + k];
    sq += part[((size_t)t * 12 + 6 + hh) * nblk + k];
  }
  float inv = 1.0f / (float)btot;
  float mu  = s * inv;
  float var = sq * inv - mu * mu;
  float sc  = gamma[idx] * rsqrtf(var + 1e-5f);
  stats[idx * 2]     = sc;
  stats[idx * 2 + 1] = beta[idx] - mu * sc;
}

__global__ __launch_bounds__(256)
void level_apply_kernel(const float* __restrict__ z, const float* __restrict__ stats,
                        const float* __restrict__ a1W, const float* __restrict__ a1b,
                        const float* __restrict__ a2W, const float* __restrict__ a2b,
                        float* __restrict__ h, float* __restrict__ out,
                        int n, int ocol)
{
  const int t = blockIdx.x;
  const int b = blockIdx.y * 256 + threadIdx.x;
  const float* zp = z + ((size_t)b * n + t) * 6;
  float* hp = h + ((size_t)b * n + t) * 6;
  float a = a1b[t];
#pragma unroll
  for (int hh = 0; hh < 6; ++hh) {
    float hv = zp[hh] * stats[(t * 6 + hh) * 2] + stats[(t * 6 + hh) * 2 + 1];
    hp[hh] = hv;
    a += hv * a1W[t * 6 + hh];
  }
  float a2 = tanhf(a) * a2W[t] + a2b[t];
  out[(size_t)b * 86 + ocol + t] = a2;
}

__global__ __launch_bounds__(256)
void final_f_kernel(const float* __restrict__ root, const float* __restrict__ fW,
                    const float* __restrict__ fb, float* __restrict__ f,
                    float* __restrict__ part, int nblk)
{
  const int tid = threadIdx.x;
  const int b   = blockIdx.x * 256 + tid;
  __shared__ float red[12 * 256];
  const float* rp = root + (size_t)b * 6;
#pragma unroll
  for (int hh = 0; hh < 6; ++hh) {
    float s = fb[hh];
#pragma unroll
    for (int i = 0; i < 6; ++i) s += rp[i] * fW[hh * 6 + i];
    float fv = tanhf(s);
    f[(size_t)b * 6 + hh] = fv;
    red[hh * 256 + tid]       = fv;
    red[(6 + hh) * 256 + tid] = fv * fv;
  }
  __syncthreads();
  for (int s = 128; s > 0; s >>= 1) {
    if (tid < s)
#pragma unroll
      for (int a = 0; a < 12; ++a) red[a * 256 + tid] += red[a * 256 + tid + s];
    __syncthreads();
  }
  if (tid < 12) part[(size_t)tid * nblk + blockIdx.x] = red[tid * 256];
}

__global__ __launch_bounds__(256)
void final_out_kernel(const float* __restrict__ f, const float* __restrict__ stats,
                      const float* __restrict__ fxW, const float* __restrict__ fxb,
                      const float* __restrict__ foW, const float* __restrict__ fob,
                      float* __restrict__ out)
{
  const int b = blockIdx.x * 256 + threadIdx.x;
  float a = fxb[0];
#pragma unroll
  for (int hh = 0; hh < 6; ++hh)
    a += (f[(size_t)b * 6 + hh] * stats[hh * 2] + stats[hh * 2 + 1]) * fxW[hh];
  float p = tanhf(a) * foW[0] + fob[0];
  out[(size_t)b * 86 + 85] = 1.0f / (1.0f + expf(-p));
}

// ---------------------------------------------------------------------------
extern "C" void kernel_launch(void* const* d_in, const int* in_sizes, int n_in,
                              void* d_out, int out_size, void* d_ws, size_t ws_size,
                              hipStream_t stream)
{
  const float* x        = (const float*)d_in[0];
  const float* gene_W   = (const float*)d_in[1];   // (1536, 4096)
  const float* gene_b   = (const float*)d_in[2];
  const float* term_W   = (const float*)d_in[3];
  const float* term_b   = (const float*)d_in[4];
  const float* bn_gamma = (const float*)d_in[5];
  const float* bn_beta  = (const float*)d_in[6];
  const float* aux1_W   = (const float*)d_in[7];
  const float* aux1_b   = (const float*)d_in[8];
  const float* aux2_W   = (const float*)d_in[9];
  const float* aux2_b   = (const float*)d_in[10];
  const float* final_W  = (const float*)d_in[11];
  const float* final_b  = (const float*)d_in[12];
  const float* fbn_g    = (const float*)d_in[13];
  const float* fbn_b    = (const float*)d_in[14];
  const float* faux_W   = (const float*)d_in[15];
  const float* faux_b   = (const float*)d_in[16];
  const float* fout_W   = (const float*)d_in[17];
  const float* fout_b   = (const float*)d_in[18];
  float* out = (float*)d_out;

  float* ws    = (float*)d_ws;
  float* h_in  = ws;                                  // B*1536
  float* zbuf  = h_in + (size_t)BATCH * NCOLS;        // B*64*6
  float* hbuf  = zbuf + (size_t)BATCH * 64 * 6;       // B*64*6
  float* part  = hbuf + (size_t)BATCH * 64 * 6;       // 64*12*NBLK
  float* stats = part + (size_t)64 * 12 * NBLK;       // 85*6*2
  float* fbuf  = stats + 85 * 6 * 2;                  // B*6
  float* fpart = fbuf + (size_t)BATCH * 6;            // 12*NBLK
  float* fstat = fpart + 12 * NBLK;                   // 12
  const size_t float_region = (size_t)(fstat + 12 - ws);   // elements

  const size_t xbf_elems = (size_t)BATCH * KDIM;
  const size_t wbf_elems = (size_t)NCOLS * KDIM;
  const size_t need = float_region * 4 + (xbf_elems + wbf_elems) * 2;

  dim3 gg(BATCH / TILE_M, NCOLS / TILE_N);
  if (ws_size >= need) {
    ushort_t* xbf = (ushort_t*)((char*)d_ws + float_region * 4);
    ushort_t* wbf = xbf + xbf_elems;
    cvt_f32_bf16_kernel<<<(int)(xbf_elems / 8 / 256), 256, 0, stream>>>(
        x, (unsigned*)xbf, (int)(xbf_elems / 8));
    cvt_f32_bf16_kernel<<<(int)(wbf_elems / 8 / 256), 256, 0, stream>>>(
        gene_W, (unsigned*)wbf, (int)(wbf_elems / 8));
    gemm_bf16_kernel<<<gg, 256, 0, stream>>>(xbf, wbf, gene_b, h_in);
  } else {
    gemm_xw_kernel<<<gg, 256, 0, stream>>>(x, gene_W, gene_b, h_in);
  }

  const int sizes[4] = {64, 16, 4, 1};
  const int offs[4]  = {0, 64, 80, 84};
  const float* in = h_in;
  for (int L = 0; L < 4; ++L) {
    int n = sizes[L], o = offs[L];
    dim3 gz(n, NBLK);
    level_z_kernel<<<gz, 256, 0, stream>>>(in, term_W + (size_t)o * 144,
                                           term_b + (size_t)o * 6, zbuf, part, n, NBLK);
    level_stats_kernel<<<(n * 6 + 63) / 64, 64, 0, stream>>>(
        part, bn_gamma + (size_t)o * 6, bn_beta + (size_t)o * 6, stats, n, NBLK, BATCH);
    level_apply_kernel<<<gz, 256, 0, stream>>>(
        zbuf, stats, aux1_W + (size_t)o * 6, aux1_b + o, aux2_W + o, aux2_b + o,
        hbuf, out, n, o);
    in = hbuf;
  }

  final_f_kernel<<<NBLK, 256, 0, stream>>>(hbuf, final_W, final_b, fbuf, fpart, NBLK);
  level_stats_kernel<<<1, 64, 0, stream>>>(fpart, fbn_g, fbn_b, fstat, 1, NBLK, BATCH);
  final_out_kernel<<<NBLK, 256, 0, stream>>>(fbuf, fstat, faux_W, faux_b, fout_W, fout_b, out);
}
